// FusedSequenceParallelAll2AllAttn_7292854469128
// MI455X (gfx1250) — compile-verified, matrix operand order checked by reading, not run on hardware
//
#include <hip/hip_runtime.h>

typedef __attribute__((ext_vector_type(16))) __bf16 v16bf;
typedef __attribute__((ext_vector_type(8)))  __bf16 v8bf;
typedef __attribute__((ext_vector_type(8)))  float  v8f;
typedef __attribute__((ext_vector_type(4)))  float  v4f;
typedef __attribute__((ext_vector_type(4)))  unsigned u32x4;
typedef unsigned short u16;

#define S_LEN   4096
#define NQH     16
#define NKVH    4
#define DH      128
#define PACKH   24          // QH + 2*KVH
#define TILE_M  64
#define TILE_N  32
#define QSCALE  0.08838834764831845f   // 1/sqrt(128)
#define NEGINF  -1.0e30f

// shard position of natural token n (zigzag, WORLD=8, L=256)
__device__ __forceinline__ int inv_zig(int n) {
    int c = n >> 8;         // chunk 0..15
    int w = n & 255;
    return (c < 8) ? (c * 512 + w) : ((15 - c) * 512 + 256 + w);
}

// hardware RNE conversions (lowers to v_cvt_pk_bf16_f32 / v_cvt_bf16_f32)
__device__ __forceinline__ unsigned pk2bf(float x, float y) {
    union { __bf16 h[2]; unsigned u; } r;
    r.h[0] = (__bf16)x; r.h[1] = (__bf16)y;
    return r.u;
}
__device__ __forceinline__ u16 bf1(float x) {
    union { __bf16 h; u16 u; } r;
    r.h = (__bf16)x;
    return r.u;
}

// load 16 bf16 in WMMA 16-bit operand layout: 8 at p, 8 at p+16 (elements)
__device__ __forceinline__ v16bf frag_ld(const u16* p) {
    const v8bf* pl = (const v8bf*)__builtin_assume_aligned(p, 16);
    const v8bf* ph = (const v8bf*)__builtin_assume_aligned(p + 16, 16);
    v8bf lo = *pl;
    v8bf hi = *ph;
    v16bf r;
#pragma unroll
    for (int i = 0; i < 8; ++i) { r[i] = lo[i]; r[i + 8] = hi[i]; }
    return r;
}

// pack 8 floats -> 8 bf16 (16B) and store to LDS as one ds_store_b128
__device__ __forceinline__ void st8bf(u16* dst, v4f a, v4f b) {
    u32x4 w;
    w.x = pk2bf(a.x, a.y);
    w.y = pk2bf(a.z, a.w);
    w.z = pk2bf(b.x, b.y);
    w.w = pk2bf(b.z, b.w);
    *(u32x4*)__builtin_assume_aligned(dst, 16) = w;
}

__global__ __launch_bounds__(128)
void zz_flash_attn_kernel(const float* __restrict__ qkv, float* __restrict__ out) {
    const int qt   = blockIdx.x;        // 0..63 natural q-tile
    const int h    = blockIdx.y;        // 0..15 query head
    const int kvh  = h >> 2;            // GQA group
    const int tid  = threadIdx.x;
    const int lane = tid & 31;
    const int wave = tid >> 5;
    const int l15  = lane & 15;
    const int hi16 = lane >> 4;

    __shared__ __align__(16) u16 Qs[TILE_M * DH];        // 16 KB bf16 row-major
    __shared__ __align__(16) u16 Ks[TILE_N * DH];        //  8 KB bf16 row-major
    __shared__ __align__(16) u16 Vs[DH * TILE_N];        //  8 KB bf16 [d][k]
    __shared__ __align__(16) u16 Ps[4 * 16 * TILE_N];    //  4 KB per-wave P tiles

    // ---- stage Q (pre-scaled) : 64 rows x 128, b128 LDS stores ----
    {
        int r  = tid >> 1;
        int dh = (tid & 1) * 64;
        int qn = qt * TILE_M + r;
        long base = ((long)inv_zig(qn) * PACKH + h) * DH + dh;
#pragma unroll
        for (int d = 0; d < 64; d += 8) {
            v4f a = *(const v4f*)(qkv + base + d);
            v4f b = *(const v4f*)(qkv + base + d + 4);
            a *= QSCALE; b *= QSCALE;
            st8bf(&Qs[r * DH + dh + d], a, b);
        }
    }
    __syncthreads();

    // per-wave Q fragments: rows wave*16 .. +15, 4 chunks of K=32
    v16bf qf[4];
    {
        const u16* qrow = &Qs[(wave * 16 + l15) * DH];
        int doff = hi16 * 8;
#pragma unroll
        for (int c = 0; c < 4; ++c) qf[c] = frag_ld(qrow + c * 32 + doff);
    }

    v8f acc[8] = {};
    float mrow[8], lrow[8];
#pragma unroll
    for (int j = 0; j < 8; ++j) { mrow[j] = -3.0e38f; lrow[j] = 0.0f; }

    const int nblocks     = 2 * qt + 2;          // causal: keys <= tile end
    const int q_wave_base = qt * TILE_M + wave * 16;

    for (int kb = 0; kb < nblocks; ++kb) {
        const int k0 = kb * TILE_N;
        __syncthreads();                         // protect Ks/Vs reuse

        // ---- stage K row-major (b128 stores) ----
        {
            int kk = tid >> 2;
            int dh = (tid & 3) * 32;
            int sp = inv_zig(k0 + kk);
            long kbase = ((long)sp * PACKH + NQH + kvh) * DH + dh;
            if (kb + 1 < nblocks) {
                long pb = ((long)inv_zig(k0 + TILE_N + kk) * PACKH + NQH + kvh) * DH + dh;
                __builtin_prefetch(qkv + pb, 0, 1);   // global_prefetch_b8
            }
#pragma unroll
            for (int d = 0; d < 32; d += 8) {
                v4f a = *(const v4f*)(qkv + kbase + d);
                v4f b = *(const v4f*)(qkv + kbase + d + 4);
                st8bf(&Ks[kk * DH + dh + d], a, b);
            }
        }
        // ---- stage V transposed [d][k]: thread owns a key PAIR x 16 d's,
        //      packs (v[2kp], v[2kp+1]) -> one b32 store per d ----
        {
            int kp = tid >> 3;            // 0..15 -> keys 2kp, 2kp+1
            int dg = (tid & 7) * 16;      // 16 d values
            long v0 = ((long)inv_zig(k0 + 2 * kp)     * PACKH + NQH + NKVH + kvh) * DH + dg;
            long v1 = ((long)inv_zig(k0 + 2 * kp + 1) * PACKH + NQH + NKVH + kvh) * DH + dg;
#pragma unroll
            for (int d = 0; d < 16; d += 4) {
                v4f a = *(const v4f*)(qkv + v0 + d);
                v4f b = *(const v4f*)(qkv + v1 + d);
                *(unsigned*)&Vs[(dg + d + 0) * TILE_N + 2 * kp] = pk2bf(a.x, b.x);
                *(unsigned*)&Vs[(dg + d + 1) * TILE_N + 2 * kp] = pk2bf(a.y, b.y);
                *(unsigned*)&Vs[(dg + d + 2) * TILE_N + 2 * kp] = pk2bf(a.z, b.z);
                *(unsigned*)&Vs[(dg + d + 3) * TILE_N + 2 * kp] = pk2bf(a.w, b.w);
            }
        }
        __syncthreads();

        // ---- S = Q K^T : two 16x16 tiles (key halves), 4 chained WMMAs each ----
        v8f s0 = {}, s1 = {};
        {
            int doff = hi16 * 8;
#pragma unroll
            for (int c = 0; c < 4; ++c) {
                v16bf kf0 = frag_ld(&Ks[(l15)      * DH + c * 32 + doff]);
                s0 = __builtin_amdgcn_wmma_f32_16x16x32_bf16(false, qf[c], false, kf0,
                                                             (short)0, s0, false, false);
                v16bf kf1 = frag_ld(&Ks[(16 + l15) * DH + c * 32 + doff]);
                s1 = __builtin_amdgcn_wmma_f32_16x16x32_bf16(false, qf[c], false, kf1,
                                                             (short)0, s1, false, false);
            }
        }

        // ---- causal mask (C layout: row = j + 8*hi16, col = l15) ----
        if (k0 + TILE_N - 1 > q_wave_base) {
#pragma unroll
            for (int j = 0; j < 8; ++j) {
                int qn = q_wave_base + j + 8 * hi16;
                if (k0 + l15      > qn) s0[j] = NEGINF;
                if (k0 + 16 + l15 > qn) s1[j] = NEGINF;
            }
        }

        // ---- online softmax (reductions stay inside 16-lane halves) ----
#pragma unroll
        for (int j = 0; j < 8; ++j) {
            float mx = fmaxf(s0[j], s1[j]);
            mx = fmaxf(mx, __shfl_xor(mx, 1));
            mx = fmaxf(mx, __shfl_xor(mx, 2));
            mx = fmaxf(mx, __shfl_xor(mx, 4));
            mx = fmaxf(mx, __shfl_xor(mx, 8));
            float mnew  = fmaxf(mrow[j], mx);
            float alpha = __expf(mrow[j] - mnew);
            float p0 = __expf(s0[j] - mnew);
            float p1 = __expf(s1[j] - mnew);
            float rs = p0 + p1;
            rs += __shfl_xor(rs, 1);
            rs += __shfl_xor(rs, 2);
            rs += __shfl_xor(rs, 4);
            rs += __shfl_xor(rs, 8);
            lrow[j] = lrow[j] * alpha + rs;
            mrow[j] = mnew;
            s0[j] = p0; s1[j] = p1;
#pragma unroll
            for (int t = 0; t < 8; ++t) acc[t][j] *= alpha;
        }

        // ---- P: C-layout -> A-layout via per-wave LDS, then acc += P V ----
        {
            u16* pw = &Ps[wave * (16 * TILE_N)];
#pragma unroll
            for (int j = 0; j < 8; ++j) {
                int row = j + 8 * hi16;
                pw[row * TILE_N + l15]      = bf1(s0[j]);
                pw[row * TILE_N + 16 + l15] = bf1(s1[j]);
            }
            // same-wave LDS ops are in-order; force the wait + block reordering
            asm volatile("s_wait_dscnt 0" ::: "memory");
            v16bf pf = frag_ld(&pw[l15 * TILE_N + hi16 * 8]);
#pragma unroll
            for (int t = 0; t < 8; ++t) {
                v16bf vf = frag_ld(&Vs[(t * 16 + l15) * TILE_N + hi16 * 8]);
                acc[t] = __builtin_amdgcn_wmma_f32_16x16x32_bf16(false, pf, false, vf,
                                                                 (short)0, acc[t], false, false);
            }
        }
    }

    // ---- epilogue: divide by softmax sum, scatter to shard layout ----
#pragma unroll
    for (int j = 0; j < 8; ++j) {
        int qn = q_wave_base + j + 8 * hi16;
        long obase = ((long)inv_zig(qn) * NQH + h) * DH;
        float rinv = 1.0f / lrow[j];
#pragma unroll
        for (int t = 0; t < 8; ++t)
            out[obase + t * 16 + l15] = acc[t][j] * rinv;
    }
}

extern "C" void kernel_launch(void* const* d_in, const int* in_sizes, int n_in,
                              void* d_out, int out_size, void* d_ws, size_t ws_size,
                              hipStream_t stream) {
    const float* qkv = (const float*)d_in[0];   // (1, 4096, 24, 128) f32, shard layout
    float* out = (float*)d_out;                 // (1, 4096, 16, 128) f32, shard layout
    dim3 grid(S_LEN / TILE_M, NQH);
    zz_flash_attn_kernel<<<grid, dim3(128), 0, stream>>>(qkv, out);
}
